// MMU_54262616818001
// MI455X (gfx1250) — matrix operand order repaired and checked by reading, not compiled, not run on hardware
//
#include <hip/hip_runtime.h>

// ---------------------------------------------------------------------------
// CDNA5 (gfx1250) fused MMU-cell forward.
// 9 live GEMMs [4096x2048]x[2048x2048] via v_wmma_f32_16x16x32_f16 (wave32),
// fp32->f16 conversion while staging tiles into LDS, fp32 accumulate.
// BK=64 staging: 16 WMMAs per barrier round, explicit global address space.
// ---------------------------------------------------------------------------

typedef __attribute__((ext_vector_type(16))) _Float16     v16h;
typedef __attribute__((ext_vector_type(2)))  _Float16     h2;
typedef __attribute__((ext_vector_type(8)))  float        v8f;
typedef __attribute__((ext_vector_type(4)))  float        f32x4;
typedef __attribute__((ext_vector_type(4)))  unsigned int u32x4;
typedef __attribute__((ext_vector_type(2)))  unsigned int u32x2;

#define BM 128
#define BN 128
#define BK 64
#define LDS_STRIDE 72   // ushorts per row (144 B): 16B-aligned rows, conflict-free

// Force global (addrspace 1) memory ops so codegen emits global_load/global_store,
// not flat_* (flat ties up DScnt as well as LOADcnt). ext_vector types are
// trivial builtin vectors, so addrspace(1) copies are plain loads.
typedef const __attribute__((address_space(1))) f32x4* gcf4;
typedef const __attribute__((address_space(1))) float* gcf;
typedef       __attribute__((address_space(1))) float* gf;

__device__ __forceinline__ f32x4 gload4(const float* p) {
    return *(gcf4)(unsigned long long)p;
}
__device__ __forceinline__ float gload(const float* p) {
    return *(gcf)(unsigned long long)p;
}
__device__ __forceinline__ void gstore(float* p, float v) {
    *(gf)(unsigned long long)p = v;
}

// Stage a 128x64 fp32 tile (rows row0.., cols k0..k0+63) into LDS as f16.
__device__ __forceinline__ void load_tile(const float* __restrict__ src,
                                          int row0, int ld, int k0,
                                          unsigned short* dst, int tid, bool pf) {
    const int cg = tid & 15;       // 16 column groups of 4 floats -> 64 cols
    int r = tid >> 4;              // 0..15, stepped by 16 (8 iters -> 128 rows)
#pragma unroll
    for (int it = 0; it < 8; ++it, r += 16) {
        const float* gp = src + (long)(row0 + r) * ld + k0 + cg * 4;
        f32x4 v = gload4(gp);                         // global_load_b128
        if (pf) __builtin_prefetch(gp + BK, 0, 0);    // next K tile prefetch
        union { h2 h[2]; u32x2 u; } pk;
        pk.h[0] = h2{ (_Float16)v.x, (_Float16)v.y }; // v_cvt_pk_f16_f32
        pk.h[1] = h2{ (_Float16)v.z, (_Float16)v.w }; // v_cvt_pk_f16_f32
        *(u32x2*)(dst + r * LDS_STRIDE + cg * 4) = pk.u;  // ds_store_b64
    }
}

// A-matrix 16x32 f16 fragment (K-half h of a 64-wide row).
// Lane l(0..15): K=0..7 & 16..23 of row l; lane l+16: K=8..15 & 24..31.
__device__ __forceinline__ v16h load_frag_a(const unsigned short* tile, int row, int sel, int h) {
    const char* p = (const char*)tile + row * (LDS_STRIDE * 2) + h * 64 + sel * 16;
    union { u32x4 q[2]; v16h f; } u;
    u.q[0] = *(const u32x4*)p;          // ds_load_b128
    u.q[1] = *(const u32x4*)(p + 32);   // ds_load_b128
    return u.f;
}

// B-matrix 32x16 f16 fragment. Lane n(0..15): K=0..7 & 16..23 of col n;
// lane n+16: K=8..15 & 24..31 of col n. (Staged W tile rows are B columns.)
__device__ __forceinline__ v16h load_frag_b(const unsigned short* tile, int row, int sel, int h) {
    const char* p = (const char*)tile + row * (LDS_STRIDE * 2) + h * 64 + sel * 16;
    union { u32x4 q[2]; v16h f; } u;
    u.q[0] = *(const u32x4*)p;
    u.q[1] = *(const u32x4*)(p + 32);
    return u.f;
}

struct GemmArgs {
    const float* A0; const float* A1; const float* A2;   // A segments [M,K]
    const float* W0; const float* W1; const float* W2;   // W segments [N,K] (nn.Linear weight)
    const float* b0; const float* b1; const float* b2;   // biases [N]
    const float* add;                                    // optional addend [M,N]
    float* C;                                            // output [M,N]
    int nseg; int nbias;
};

__global__ __launch_bounds__(256)
void mmu_gemm_wmma(GemmArgs g) {
    constexpr int N = 2048, K = 2048;
    __shared__ unsigned short As[BM * LDS_STRIDE];  // 18 KB
    __shared__ unsigned short Bs[BN * LDS_STRIDE];  // 18 KB

    const int tid  = threadIdx.x;
    const int lane = tid & 31;
    const int wave = tid >> 5;          // 8 waves: 2 (M) x 4 (N)
    const int wm = wave >> 2, wn = wave & 3;
    const int ln = lane & 15, sel = lane >> 4;
    const int m0 = blockIdx.x * BM, n0 = blockIdx.y * BN;

    const float* Aseg[3] = { g.A0, g.A1, g.A2 };
    const float* Wseg[3] = { g.W0, g.W1, g.W2 };

    v8f acc[4][2] = {};   // wave tile 64x32 = 4x2 WMMA tiles

    for (int s = 0; s < g.nseg; ++s) {
        const float* A = Aseg[s];
        const float* W = Wseg[s];
        for (int k0 = 0; k0 < K; k0 += BK) {
            __syncthreads();
            load_tile(A, m0, K, k0, As, tid, k0 + BK < K);
            load_tile(W, n0, K, k0, Bs, tid, k0 + BK < K);
            __syncthreads();
#pragma unroll
            for (int h = 0; h < 2; ++h) {
                v16h af[4], bf[2];
#pragma unroll
                for (int tm = 0; tm < 4; ++tm)
                    af[tm] = load_frag_a(As, wm * 64 + tm * 16 + ln, sel, h);
#pragma unroll
                for (int tn = 0; tn < 2; ++tn)
                    bf[tn] = load_frag_b(Bs, wn * 32 + tn * 16 + ln, sel, h);
#pragma unroll
                for (int tm = 0; tm < 4; ++tm)
#pragma unroll
                    for (int tn = 0; tn < 2; ++tn)
                        acc[tm][tn] = __builtin_amdgcn_wmma_f32_16x16x32_f16(
                            /*neg_a=*/false, af[tm], /*neg_b=*/false, bf[tn],
                            /*c_mod=*/(short)0, acc[tm][tn],
                            /*reuse_a=*/false, /*reuse_b=*/false);
            }
        }
    }

    // Epilogue. C/D layout: VGPR i -> lanes 0-15: (M=i, N=lane); lanes 16-31: (M=i+8, N=lane-16)
    const float* bseg[3] = { g.b0, g.b1, g.b2 };
    const bool has_add = (g.add != nullptr);
#pragma unroll
    for (int tn = 0; tn < 2; ++tn) {
        const int col = n0 + wn * 32 + tn * 16 + ln;
        float bsum = 0.f;
        for (int j = 0; j < g.nbias; ++j) bsum += gload(bseg[j] + col);
#pragma unroll
        for (int tm = 0; tm < 4; ++tm) {
            const int rbase = m0 + wm * 64 + tm * 16 + sel * 8;
#pragma unroll
            for (int i = 0; i < 8; ++i) {
                const long idx = (long)(rbase + i) * N + col;
                float v = acc[tm][tn][i] + bsum;
                if (has_add) v += gload(g.add + idx);
                gstore(g.C + idx, v);
            }
        }
    }
}

// gate_prod = sigmoid(p_inp)*sigmoid(p_ig);  rg_mem = sigmoid(p_rg)*mem0
__global__ __launch_bounds__(256)
void mmu_gates_ew(const float* __restrict__ pinp, const float* __restrict__ pig,
                  const float* __restrict__ prg, const float* __restrict__ mem0,
                  float* gp, float* rm, int n) {
    int i = blockIdx.x * blockDim.x + threadIdx.x;
    if (i < n) {
        float a = gload(pinp + i), b = gload(pig + i);
        float c = gload(prg + i),  m = gload(mem0 + i);
        float sa = 1.f / (1.f + __expf(-a));
        float sb = 1.f / (1.f + __expf(-b));
        float sc = 1.f / (1.f + __expf(-c));
        gstore(gp + i, sa * sb);   // in-place over pinp buffer: same index, safe
        gstore(rm + i, sc * m);    // in-place over prg buffer
    }
}

extern "C" void kernel_launch(void* const* d_in, const int* in_sizes, int n_in,
                              void* d_out, int out_size, void* d_ws, size_t ws_size,
                              hipStream_t stream) {
    (void)in_sizes; (void)n_in; (void)out_size; (void)ws_size;
    const float* x    = (const float*)d_in[0];
    const float* out0 = (const float*)d_in[1];
    const float* mem0 = (const float*)d_in[2];
    const float* w_inpgate      = (const float*)d_in[3];
    const float* b_inpgate      = (const float*)d_in[4];
    const float* w_rec_inpgate  = (const float*)d_in[5];
    const float* b_rec_inpgate  = (const float*)d_in[6];
    const float* w_mem_inpgate  = (const float*)d_in[7];
    const float* b_mem_inpgate  = (const float*)d_in[8];
    const float* w_inp          = (const float*)d_in[9];
    const float* b_inp          = (const float*)d_in[10];
    const float* w_rec_inp      = (const float*)d_in[11];
    const float* b_rec_inp      = (const float*)d_in[12];
    const float* w_readgate     = (const float*)d_in[13];
    const float* b_readgate     = (const float*)d_in[14];
    const float* w_rec_readgate = (const float*)d_in[15];
    const float* b_rec_readgate = (const float*)d_in[16];
    const float* w_mem_readgate = (const float*)d_in[17];
    const float* b_mem_readgate = (const float*)d_in[18];
    const float* w_decoder      = (const float*)d_in[19];
    const float* b_decoder      = (const float*)d_in[20];
    // writegate / encoder weights are dead for the returned output.

    const size_t SZ = (size_t)4096 * 2048;
    float* P0 = (float*)d_ws;       // preact_inp      -> gate_prod (in place)
    float* P1 = P0 + SZ;            // preact_inpgate
    float* P2 = P1 + SZ;            // preact_readgate -> rg_mem (in place)

    dim3 grid(4096 / BM, 2048 / BN), block(256);

    GemmArgs g;
    // preact_inp = x W_inp^T + out0 W_rec_inp^T + b_inp + b_rec_inp
    g = { x, out0, nullptr, w_inp, w_rec_inp, nullptr,
          b_inp, b_rec_inp, nullptr, nullptr, P0, 2, 2 };
    mmu_gemm_wmma<<<grid, block, 0, stream>>>(g);

    // preact_inpgate = x W_ig^T + mem0 W_mig^T + out0 W_rig^T + biases
    g = { x, mem0, out0, w_inpgate, w_mem_inpgate, w_rec_inpgate,
          b_inpgate, b_mem_inpgate, b_rec_inpgate, nullptr, P1, 3, 3 };
    mmu_gemm_wmma<<<grid, block, 0, stream>>>(g);

    // preact_readgate
    g = { x, mem0, out0, w_readgate, w_mem_readgate, w_rec_readgate,
          b_readgate, b_mem_readgate, b_rec_readgate, nullptr, P2, 3, 3 };
    mmu_gemm_wmma<<<grid, block, 0, stream>>>(g);

    // gates + read-gated memory
    const int n = (int)SZ;
    mmu_gates_ew<<<(n + 255) / 256, 256, 0, stream>>>(P0, P1, P2, mem0, P0, P2, n);

    // hidden = rg_mem W_dec^T + b_dec + gate_prod  -> d_out
    g = { P2, nullptr, nullptr, w_decoder, nullptr, nullptr,
          b_decoder, nullptr, nullptr, P0, (float*)d_out, 1, 1 };
    mmu_gemm_wmma<<<grid, block, 0, stream>>>(g);
}
